// custom_reg_head_68375879352702
// MI455X (gfx1250) — matrix operand order repaired and checked by reading, not compiled, not run on hardware
//
#include <hip/hip_runtime.h>

typedef __attribute__((ext_vector_type(16))) _Float16 v16h;
typedef __attribute__((ext_vector_type(8)))  _Float16 v8h;
typedef __attribute__((ext_vector_type(8)))  float    v8f;
typedef __attribute__((ext_vector_type(4)))  float    f32x4;

#define D_DIM    4096
#define TWO_D    8192
#define KSPLITS  4
#define RGROUPS  2          // row-groups of 16 rows per block
#define BLOCK    256        // 8 waves: 2 row-groups x 4 K-splits
#define ROWS_PER_BLOCK 32

// y[i] = sigmoid( dot( out2d[i, t[i]*D : t[i]*D+D], W ) + b )
// GEMV done as WMMA f16 16x16x32 with W broadcast across all 16 B-columns.
__global__ __launch_bounds__(BLOCK)
void sel_gemv_sigmoid_wmma(const float* __restrict__ out2d,  // [B, 2D] f32
                           const int*   __restrict__ t,      // [B]
                           const float* __restrict__ W,      // [D]  f32
                           const float* __restrict__ bias,   // [1]
                           float*       __restrict__ y)      // [B]
{
    __shared__ __align__(16) _Float16 Wh[D_DIM];                 // 8 KiB f16 copy of W
    __shared__ float partial[KSPLITS * RGROUPS * 16];            // 512 B

    const int tid = threadIdx.x;

    // Convert W -> f16 once per block into LDS.
    for (int i = tid; i < D_DIM; i += BLOCK)
        Wh[i] = (_Float16)W[i];
    __syncthreads();

    const int wave     = tid >> 5;
    const int lane     = tid & 31;
    const int kSplit   = wave & 3;        // 4-way split of K
    const int rowGroup = wave >> 2;       // 2 groups of 16 rows
    const int m        = lane & 15;       // A-matrix row within the 16-row tile
    const int hi       = lane >> 4;       // 0: K 0-7,16-23 ; 1: K 8-15,24-31

    const int    row   = blockIdx.x * ROWS_PER_BLOCK + rowGroup * 16 + m;
    const size_t rbase = (size_t)row * TWO_D + (t[row] ? (size_t)D_DIM : 0);
    const float* __restrict__ A = out2d + rbase;

    const int k_begin = kSplit * (D_DIM / KSPLITS);
    const int k_end   = k_begin + (D_DIM / KSPLITS);
    const int koff    = hi * 8;

    v8f c = {};
    #pragma unroll 2
    for (int k0 = k_begin; k0 < k_end; k0 += 32) {
        // --- A tile: 16 rows x 32 K, f32 stream -> f16 registers ---
        // lane<16 needs K k0+{0..7, 16..23}; lane>=16 needs K k0+{8..15, 24..31}
        const f32x4 r0 = __builtin_nontemporal_load((const f32x4*)(A + k0 + koff));
        const f32x4 r1 = __builtin_nontemporal_load((const f32x4*)(A + k0 + koff + 4));
        const f32x4 r2 = __builtin_nontemporal_load((const f32x4*)(A + k0 + koff + 16));
        const f32x4 r3 = __builtin_nontemporal_load((const f32x4*)(A + k0 + koff + 20));
        v16h a;
        #pragma unroll
        for (int j = 0; j < 4; ++j) {
            a[j]      = (_Float16)r0[j];
            a[4 + j]  = (_Float16)r1[j];
            a[8 + j]  = (_Float16)r2[j];
            a[12 + j] = (_Float16)r3[j];
        }

        // --- B tile: W[k0..k0+31] broadcast to all 16 columns ---
        // lanes 0-15 hold K k0+{0..7, 16..23}, lanes 16-31 hold K k0+{8..15, 24..31}
        const v8h b0 = *(const v8h*)(Wh + k0 + hi * 8);
        const v8h b1 = *(const v8h*)(Wh + k0 + hi * 8 + 16);
        v16h bt;
        #pragma unroll
        for (int j = 0; j < 8; ++j) { bt[j] = b0[j]; bt[8 + j] = b1[j]; }

        // D = A x B + C  (every column of D holds the same dot product)
        c = __builtin_amdgcn_wmma_f32_16x16x32_f16(
                /*neg_a=*/false, a, /*neg_b=*/false, bt,
                /*c_mod=*/(short)0, c, /*reuse_a=*/false, /*reuse_b=*/false);
    }

    // Column N==0 lanes (lane 0 -> rows M=0..7, lane 16 -> rows M=8..15)
    // publish this wave's 16 partial dot products.
    if ((lane & 15) == 0) {
        float* p = &partial[(kSplit * RGROUPS + rowGroup) * 16 + hi * 8];
        #pragma unroll
        for (int j = 0; j < 8; ++j) p[j] = c[j];
    }
    __syncthreads();

    // Reduce the 4 K-splits, add bias, sigmoid, store (32 rows per block).
    if (tid < ROWS_PER_BLOCK) {
        const int rg = tid >> 4, mm = tid & 15;
        float s = bias[0];
        #pragma unroll
        for (int ks = 0; ks < KSPLITS; ++ks)
            s += partial[(ks * RGROUPS + rg) * 16 + mm];
        y[blockIdx.x * ROWS_PER_BLOCK + tid] = 1.0f / (1.0f + __expf(-s));
    }
}

extern "C" void kernel_launch(void* const* d_in, const int* in_sizes, int n_in,
                              void* d_out, int out_size, void* d_ws, size_t ws_size,
                              hipStream_t stream) {
    (void)n_in; (void)out_size; (void)d_ws; (void)ws_size;
    const float* out2d = (const float*)d_in[0];   // [B, 2D]
    const int*   t     = (const int*)d_in[1];     // [B]
    const float* W     = (const float*)d_in[2];   // [1, D]
    const float* bias  = (const float*)d_in[3];   // [1]
    float*       y     = (float*)d_out;           // [B]

    const int B = in_sizes[1];                    // 16384
    dim3 grid(B / ROWS_PER_BLOCK);                // 512 blocks x 8 waves = 4096 waves
    sel_gemv_sigmoid_wmma<<<grid, BLOCK, 0, stream>>>(out2d, t, W, bias, y);
}
